// LSTMSelector_16870631538956
// MI455X (gfx1250) — matrix operand order repaired and checked by reading, not compiled, not run on hardware
//
#include <hip/hip_runtime.h>
#include <hip/hip_bf16.h>

#define BS 128
#define NS 1024
#define NSTEP 8
#define IN_DIM 768
#define HID 512
#define MLP 512
#define GATES4 2048
#define NEGV (-100000.0f)

typedef __bf16 v16bf __attribute__((ext_vector_type(16)));
typedef float v8f __attribute__((ext_vector_type(8)));
typedef unsigned int v8u __attribute__((ext_vector_type(8)));
typedef unsigned int u32x4 __attribute__((ext_vector_type(4)));
typedef int i32x4 __attribute__((ext_vector_type(4)));
typedef int i32x8 __attribute__((ext_vector_type(8)));

#if defined(__HIP_DEVICE_COMPILE__) && defined(__has_builtin)
#if __has_builtin(__builtin_amdgcn_tensor_load_to_lds) && __has_builtin(__builtin_amdgcn_s_wait_tensorcnt)
#define HAVE_TDM 1
#endif
#endif

__device__ __forceinline__ unsigned short f2bf(float f) {
  unsigned int u = __float_as_uint(f);
  unsigned int r = u + 0x7FFFu + ((u >> 16) & 1u);
  return (unsigned short)(r >> 16);
}
__device__ __forceinline__ float bf2f(unsigned short h) {
  return __uint_as_float(((unsigned int)h) << 16);
}

__device__ __forceinline__ v8f wmma_bf16(v16bf a, v16bf b, v8f c) {
  // D = A(16x32 bf16) x B(32x16 bf16) + C(16x16 f32)
  return __builtin_amdgcn_wmma_f32_16x16x32_bf16(
      /*neg_a=*/false, a, /*neg_b=*/false, b,
      /*c_mod=*/(short)0, c, /*reuse_a=*/false, /*reuse_b=*/false);
}

// A fragment: 16(M) x 32(K) bf16, rows from row-major [rows][K] bf16 in global.
// Lane layout: M = lane%16; VGPR v holds K-pair
//   kb = 2v + (lane>=16 ? 16 : 0)
__device__ __forceinline__ v16bf load_a_g(const unsigned short* __restrict__ base,
                                          int row_base, int K, int k0, int lane) {
  v8u p;
  int m = row_base + (lane & 15);
  int hk = (lane >= 16) ? 16 : 0;
  const unsigned short* rp = base + (size_t)m * K + k0;
#pragma unroll
  for (int v = 0; v < 8; ++v) {
    int kb = 2 * v + hk;
    p[v] = *(const unsigned int*)(rp + kb);
  }
  return __builtin_bit_cast(v16bf, p);
}

// B fragment: 32(K) x 16(N); element(k,n) = W[n][k] from row-major [cols][K] bf16.
// Lane layout: N = lane%16; VGPR v holds K-pair kb = 2v + (lane>=16 ? 16 : 0)
__device__ __forceinline__ v16bf load_b_g(const unsigned short* __restrict__ W,
                                          int col_base, int K, int k0, int lane) {
  v8u p;
  int n = col_base + (lane & 15);
  const unsigned short* rp = W + (size_t)n * K + k0 + ((lane >= 16) ? 16 : 0);
#pragma unroll
  for (int v = 0; v < 8; ++v) p[v] = *(const unsigned int*)(rp + 2 * v);
  return __builtin_bit_cast(v16bf, p);
}

// B fragment from LDS, LDS layout: [16 rows][Kd dwords], row index = N = lane%16.
__device__ __forceinline__ v16bf load_b_l(const unsigned int* lds, int Kd, int k0, int lane) {
  v8u p;
  const unsigned int* rp = lds + (lane & 15) * Kd + (k0 >> 1) + ((lane >= 16) ? 8 : 0);
#pragma unroll
  for (int v = 0; v < 8; ++v) p[v] = rp[v];
  return __builtin_bit_cast(v16bf, p);
}

#ifdef HAVE_TDM
// Issue one 2-D TDM tile load: bf16 elements, tile (tile_d1 rows x tile_d0 cols),
// row stride d0_stride (elements). D# packing per CDNA5 ISA.
__device__ __forceinline__ void tdm_load_2d_bf16(const void* gaddr, unsigned int lds_off,
                                                 unsigned int tensor_d0, unsigned int tensor_d1,
                                                 unsigned int tile_d0, unsigned int tile_d1,
                                                 unsigned long long d0_stride) {
  unsigned long long ga = (unsigned long long)gaddr;
  u32x4 g0;
  g0[0] = 1u;                                                    // count=1, user mode
  g0[1] = lds_off;                                               // lds_addr (bytes)
  g0[2] = (unsigned int)ga;                                      // global_addr[31:0]
  g0[3] = (unsigned int)((ga >> 32) & 0x1FFFFFFu) | (2u << 30);  // ga[56:32], type=2
  i32x8 g1;
  g1[0] = (int)(1u << 16);                                        // data_size=1 (2 bytes)
  g1[1] = (int)((tensor_d0 & 0xFFFFu) << 16);                     // tensor_dim0[15:0]
  g1[2] = (int)((tensor_d0 >> 16) | ((tensor_d1 & 0xFFFFu) << 16));
  g1[3] = (int)((tensor_d1 >> 16) | (tile_d0 << 16));             // tile_dim0
  g1[4] = (int)tile_d1;                                           // tile_dim1 (tile_dim2=0)
  g1[5] = (int)(unsigned int)d0_stride;                           // tensor_dim0_stride[31:0]
  g1[6] = (int)((d0_stride >> 32) & 0xFFFFu);                     // stride[47:32]
  g1[7] = 0;
  i32x4 gz = {0, 0, 0, 0};
#if __clang_major__ >= 23
  i32x8 gz8 = {0, 0, 0, 0, 0, 0, 0, 0};
  __builtin_amdgcn_tensor_load_to_lds(g0, g1, gz, gz, gz8, 0);
#else
  __builtin_amdgcn_tensor_load_to_lds(g0, g1, gz, gz, 0);
#endif
}
#endif

// ---------------- setup kernels ----------------

__global__ void k_cvt(const float* __restrict__ s, unsigned short* __restrict__ d, int n) {
  int i = blockIdx.x * blockDim.x + threadIdx.x;
  if (i < n) d[i] = f2bf(s[i]);
}

__global__ void k_init(const float* __restrict__ target_emb, const int* __restrict__ ctx_len,
                       float* __restrict__ mask, float* __restrict__ c,
                       unsigned short* __restrict__ h_bf, unsigned short* __restrict__ x_bf,
                       float* __restrict__ lp) {
  int idx = blockIdx.x * blockDim.x + threadIdx.x;  // 0 .. BS*NS-1
  int b = idx >> 10, n = idx & (NS - 1);
  mask[idx] = (n >= ctx_len[b]) ? NEGV : 0.0f;
  if (idx < BS * HID) { c[idx] = 0.f; h_bf[idx] = 0; }
  if (idx < BS * IN_DIM) x_bf[idx] = f2bf(target_emb[idx]);
  if (idx < BS) lp[idx] = 0.f;
}

// ---------------- ctx_part GEMM: cp[b][m][n] = ctx[b,n,:]·W1c[m,:] + b1[m] ----------------
// grid (NS/16, MLP/128, BS), block 256 (8 waves); wave w -> m-tile, block -> 16 n's.

__global__ void k_ctxpart(const float* __restrict__ ctx, const unsigned short* __restrict__ W1c_bf,
                          const float* __restrict__ b1, unsigned short* __restrict__ cp) {
  __shared__ unsigned int ldsB[16 * (IN_DIM / 2)];  // 24 KB bf16-packed ctx tile
  int t = threadIdx.x;
  int wave = t >> 5, lane = t & 31;
  int n0 = blockIdx.x * 16;
  int m_base = (blockIdx.y * 8 + wave) * 16;
  int b = blockIdx.z;

  const float* cbase = ctx + ((size_t)b * NS + n0) * IN_DIM;
  for (int i = t; i < 16 * (IN_DIM / 2); i += 256) {
    int n = i / (IN_DIM / 2);
    int kp = i - n * (IN_DIM / 2);
    const float2 v = *((const float2*)(cbase + (size_t)n * IN_DIM) + kp);
    ldsB[i] = (unsigned int)f2bf(v.x) | ((unsigned int)f2bf(v.y) << 16);
  }
  __syncthreads();

  v8f acc = {};
#pragma unroll 4
  for (int k0 = 0; k0 < IN_DIM; k0 += 32) {
    v16bf a = load_a_g(W1c_bf, m_base, IN_DIM, k0, lane);
    v16bf bb = load_b_l(ldsB, IN_DIM / 2, k0, lane);
    acc = wmma_bf16(a, bb, acc);
  }

  int half8 = (lane >= 16) ? 8 : 0;
  int nloc = n0 + (lane & 15);
#pragma unroll
  for (int r = 0; r < 8; ++r) {
    int m = m_base + r + half8;
    float val = acc[r] + b1[m];
    cp[((size_t)b * MLP + m) * NS + nloc] = f2bf(val);
  }
}

// ---------------- LSTM gates GEMM: gates[b][col] = x·Wih^T + h·Whh^T + biases ----------------
// grid (GATES4/128, BS/16), block 256.

__global__ void k_gates(const unsigned short* __restrict__ x_bf, const unsigned short* __restrict__ h_bf,
                        const unsigned short* __restrict__ Wih, const unsigned short* __restrict__ Whh,
                        const float* __restrict__ b_ih, const float* __restrict__ b_hh,
                        float* __restrict__ gates) {
  int t = threadIdx.x, wave = t >> 5, lane = t & 31;
  int col_base = blockIdx.x * 128 + wave * 16;
  int b_base = blockIdx.y * 16;
  v8f acc = {};
#pragma unroll 4
  for (int k0 = 0; k0 < IN_DIM; k0 += 32)
    acc = wmma_bf16(load_a_g(x_bf, b_base, IN_DIM, k0, lane),
                    load_b_g(Wih, col_base, IN_DIM, k0, lane), acc);
#pragma unroll 4
  for (int k0 = 0; k0 < HID; k0 += 32)
    acc = wmma_bf16(load_a_g(h_bf, b_base, HID, k0, lane),
                    load_b_g(Whh, col_base, HID, k0, lane), acc);
  int half8 = (lane >= 16) ? 8 : 0;
  int col = col_base + (lane & 15);
  float bias = b_ih[col] + b_hh[col];
#pragma unroll
  for (int r = 0; r < 8; ++r) {
    int b = b_base + r + half8;
    gates[(size_t)b * GATES4 + col] = acc[r] + bias;
  }
}

// ---------------- LSTM cell elementwise ----------------

__global__ void k_cell(const float* __restrict__ gates, float* __restrict__ c,
                       unsigned short* __restrict__ h_bf) {
  int idx = blockIdx.x * blockDim.x + threadIdx.x;  // 0 .. BS*HID-1
  int b = idx >> 9, u = idx & (HID - 1);
  const float* g = gates + (size_t)b * GATES4;
  float ig = g[u], fg = g[HID + u], gg = g[2 * HID + u], og = g[3 * HID + u];
  float si = 1.f / (1.f + __expf(-ig));
  float sf = 1.f / (1.f + __expf(-fg));
  float so = 1.f / (1.f + __expf(-og));
  float cn = sf * c[idx] + si * tanhf(gg);
  float hn = so * tanhf(cn);
  c[idx] = cn;
  h_bf[idx] = f2bf(hn);
}

// ---------------- hproj GEMM: hproj[b][m] = h·W1h^T ----------------
// grid (MLP/128, BS/16), block 256.

__global__ void k_hproj(const unsigned short* __restrict__ h_bf, const unsigned short* __restrict__ W1h,
                        float* __restrict__ hproj) {
  int t = threadIdx.x, wave = t >> 5, lane = t & 31;
  int col_base = blockIdx.x * 128 + wave * 16;
  int b_base = blockIdx.y * 16;
  v8f acc = {};
#pragma unroll 4
  for (int k0 = 0; k0 < HID; k0 += 32)
    acc = wmma_bf16(load_a_g(h_bf, b_base, HID, k0, lane),
                    load_b_g(W1h, col_base, HID, k0, lane), acc);
  int half8 = (lane >= 16) ? 8 : 0;
  int col = col_base + (lane & 15);
#pragma unroll
  for (int r = 0; r < 8; ++r) {
    int b = b_base + r + half8;
    hproj[(size_t)b * MLP + col] = acc[r];
  }
}

// ---------------- score: sc[b][n] = sum_m tanh(cp[b][m][n]+hproj[b][m])*w2[m] + b2 + mask ---------
// grid (NS/256, BS), block 256. TDM double-buffers 32x256 bf16 tiles of cp into LDS
// (TENSORcnt-tracked) while the VALU runs the tanh-dot from the other buffer.

#define SC_MT 32                      // m rows per TDM tile
#define SC_NT (MLP / SC_MT)           // 16 tiles

__global__ void k_score(const unsigned short* __restrict__ cp, const float* __restrict__ hproj,
                        const float* __restrict__ w2, const float* __restrict__ b2,
                        const float* __restrict__ mask, float* __restrict__ sc) {
  __shared__ float hp[MLP];
  __shared__ float w2s[MLP];
  int t = threadIdx.x;
  int b = blockIdx.y;
  int n0 = blockIdx.x * 256;
  for (int i = t; i < MLP; i += 256) {
    hp[i] = hproj[(size_t)b * MLP + i];
    w2s[i] = w2[i];
  }
  const unsigned short* gbase = cp + (size_t)b * MLP * NS + n0;
  float acc = 0.f;

#ifdef HAVE_TDM
  __shared__ __align__(1024) unsigned short tile[2][SC_MT * 256];  // 2 x 16 KB
  bool issuer = (t < 32);  // wave 0 drives the TDM pipeline
  if (issuer)
    tdm_load_2d_bf16(gbase, (unsigned int)(unsigned long long)(void*)&tile[0][0],
                     NS, MLP, 256, SC_MT, NS);
  __syncthreads();  // hp/w2s ready
#pragma unroll 1
  for (int it = 0; it < SC_NT; ++it) {
    if (issuer) {
      if (it + 1 < SC_NT) {
        tdm_load_2d_bf16(gbase + (size_t)(it + 1) * SC_MT * NS,
                         (unsigned int)(unsigned long long)(void*)&tile[(it + 1) & 1][0],
                         NS, MLP, 256, SC_MT, NS);
        __builtin_amdgcn_s_wait_tensorcnt(1);  // tile `it` landed
      } else {
        __builtin_amdgcn_s_wait_tensorcnt(0);  // last tile landed
      }
    }
    __syncthreads();  // publish tile `it` to all waves
    const unsigned short* tp = &tile[it & 1][0];
    int m0 = it * SC_MT;
#pragma unroll 8
    for (int ml = 0; ml < SC_MT; ++ml) {
      float v = bf2f(tp[ml * 256 + t]) + hp[m0 + ml];
      acc += tanhf(v) * w2s[m0 + ml];
    }
    __syncthreads();  // tile `it` consumed before it is overwritten
  }
#else
  __syncthreads();
  const unsigned short* cpb = gbase + t;
#pragma unroll 4
  for (int m = 0; m < MLP; ++m) {
    if ((m & 31) == 0) __builtin_prefetch(cpb + (size_t)(m + 32) * NS, 0, 0);
    float v = bf2f(cpb[(size_t)m * NS]) + hp[m];
    acc += tanhf(v) * w2s[m];
  }
#endif

  int n = n0 + t;
  sc[(size_t)b * NS + n] = acc + b2[0] + mask[(size_t)b * NS + n];
}

// ---------------- select: gumbel-argmax, log-softmax update, mask/x update ----------------
// grid (BS), block 256.

__global__ void k_select(const float* __restrict__ sc, const float* __restrict__ gmb,
                         const float* __restrict__ ctx, float* __restrict__ mask,
                         float* __restrict__ lp, unsigned short* __restrict__ x_bf,
                         float* __restrict__ out, int step) {
  __shared__ float sv[NS];
  __shared__ float rv[256];
  __shared__ int ri[256];
  __shared__ float rm[256];
  __shared__ int s_out;
  int t = threadIdx.x, b = blockIdx.x;
  const float* scb = sc + (size_t)b * NS;
  const float* gb = gmb + ((size_t)step * BS + b) * NS;

  float bk = -3.4e38f; int bi = 0; float mx = -3.4e38f;
  for (int i = t; i < NS; i += 256) {
    float s = scb[i];
    sv[i] = s;
    float k = s + gb[i];
    if (k > bk) { bk = k; bi = i; }
    if (s > mx) mx = s;
  }
  rv[t] = bk; ri[t] = bi; rm[t] = mx;
  __syncthreads();
  for (int o = 128; o > 0; o >>= 1) {
    if (t < o) {
      if (rv[t + o] > rv[t]) { rv[t] = rv[t + o]; ri[t] = ri[t + o]; }
      if (rm[t + o] > rm[t]) rm[t] = rm[t + o];
    }
    __syncthreads();
  }
  if (t == 0) s_out = ri[0];
  float smax = rm[0];
  __syncthreads();

  float ls = 0.f;
  for (int i = t; i < NS; i += 256) ls += __expf(sv[i] - smax);
  rv[t] = ls;
  __syncthreads();
  for (int o = 128; o > 0; o >>= 1) {
    if (t < o) rv[t] += rv[t + o];
    __syncthreads();
  }

  int sel = s_out;
  if (t == 0) {
    float lse = smax + __logf(rv[0]);
    float lpn = lp[b] + sv[sel] - lse;
    lp[b] = lpn;
    out[step * BS + b] = (float)sel;
    out[NSTEP * BS + b] = lpn;  // log_probs section; final step leaves final value
    mask[(size_t)b * NS + sel] = NEGV;
  }
  const float* src = ctx + ((size_t)b * NS + sel) * IN_DIM;
  for (int i = t; i < IN_DIM; i += 256) x_bf[b * IN_DIM + i] = f2bf(src[i]);
}

// ---------------- host ----------------

extern "C" void kernel_launch(void* const* d_in, const int* in_sizes, int n_in,
                              void* d_out, int out_size, void* d_ws, size_t ws_size,
                              hipStream_t stream) {
  (void)in_sizes; (void)n_in; (void)out_size; (void)ws_size;
  const float* target_emb = (const float*)d_in[0];
  const float* ctx_emb    = (const float*)d_in[1];
  const int*   ctx_len    = (const int*)d_in[3];
  const float* W_ih       = (const float*)d_in[5];
  const float* W_hh       = (const float*)d_in[6];
  const float* b_ih       = (const float*)d_in[7];
  const float* b_hh       = (const float*)d_in[8];
  const float* W1c        = (const float*)d_in[9];
  const float* W1h        = (const float*)d_in[10];
  const float* b1         = (const float*)d_in[11];
  const float* w2         = (const float*)d_in[12];
  const float* b2         = (const float*)d_in[13];
  const float* gumbel     = (const float*)d_in[14];
  float* out = (float*)d_out;

  char* ws = (char*)d_ws;
  size_t off = 0;
  auto alloc = [&](size_t bytes) {
    void* p = ws + off;
    off = (off + bytes + 255) & ~(size_t)255;
    return p;
  };
  unsigned short* cp      = (unsigned short*)alloc((size_t)BS * MLP * NS * 2);   // 128 MB
  unsigned short* Wih_bf  = (unsigned short*)alloc((size_t)GATES4 * IN_DIM * 2);
  unsigned short* Whh_bf  = (unsigned short*)alloc((size_t)GATES4 * HID * 2);
  unsigned short* W1c_bf  = (unsigned short*)alloc((size_t)MLP * IN_DIM * 2);
  unsigned short* W1h_bf  = (unsigned short*)alloc((size_t)MLP * HID * 2);
  unsigned short* h_bf    = (unsigned short*)alloc((size_t)BS * HID * 2);
  unsigned short* x_bf    = (unsigned short*)alloc((size_t)BS * IN_DIM * 2);
  float* cbuf   = (float*)alloc((size_t)BS * HID * 4);
  float* maskb  = (float*)alloc((size_t)BS * NS * 4);
  float* lp     = (float*)alloc((size_t)BS * 4);
  float* gates  = (float*)alloc((size_t)BS * GATES4 * 4);
  float* hproj  = (float*)alloc((size_t)BS * MLP * 4);
  float* sc     = (float*)alloc((size_t)BS * NS * 4);

  k_cvt<<<dim3((GATES4 * IN_DIM + 255) / 256), 256, 0, stream>>>(W_ih, Wih_bf, GATES4 * IN_DIM);
  k_cvt<<<dim3((GATES4 * HID + 255) / 256), 256, 0, stream>>>(W_hh, Whh_bf, GATES4 * HID);
  k_cvt<<<dim3((MLP * IN_DIM + 255) / 256), 256, 0, stream>>>(W1c, W1c_bf, MLP * IN_DIM);
  k_cvt<<<dim3((MLP * HID + 255) / 256), 256, 0, stream>>>(W1h, W1h_bf, MLP * HID);
  k_init<<<dim3(BS * NS / 256), 256, 0, stream>>>(target_emb, ctx_len, maskb, cbuf, h_bf, x_bf, lp);

  k_ctxpart<<<dim3(NS / 16, MLP / 128, BS), 256, 0, stream>>>(ctx_emb, W1c_bf, b1, cp);

  for (int s = 0; s < NSTEP; ++s) {
    k_gates<<<dim3(GATES4 / 128, BS / 16), 256, 0, stream>>>(x_bf, h_bf, Wih_bf, Whh_bf, b_ih, b_hh, gates);
    k_cell<<<dim3(BS * HID / 256), 256, 0, stream>>>(gates, cbuf, h_bf);
    k_hproj<<<dim3(MLP / 128, BS / 16), 256, 0, stream>>>(h_bf, W1h_bf, hproj);
    k_score<<<dim3(NS / 256, BS), 256, 0, stream>>>(cp, hproj, w2, b2, maskb, sc);
    k_select<<<dim3(BS), 256, 0, stream>>>(sc, gumbel, ctx_emb, maskb, lp, x_bf, out, s);
  }
}
